// LSTMTagger_56581899157590
// MI455X (gfx1250) — matrix operand order repaired and checked by reading, not compiled, not run on hardware
//
#include <hip/hip_runtime.h>
#include <hip/hip_bf16.h>

// ---------------- problem constants ----------------
#define B_    64
#define P_    80
#define LC_   4
#define E_    256
#define H_    1024
#define KW_   3
#define CH_   4
#define TAGS_ 50
#define D_    260           // E_ + CH_
#define DP_   288           // D_ padded to multiple of 32 (WMMA K granule)
#define ROWS_ (B_*P_)       // 5120 flat rows
#define G4_   (4*H_)        // 4096 gate columns

typedef __attribute__((ext_vector_type(16))) __bf16 v16bf;
typedef __attribute__((ext_vector_type(8)))  float  v8f;

__device__ __forceinline__ float sigf(float x){ return 1.0f/(1.0f+__expf(-x)); }

// ---- WMMA bf16 fragment loaders (ISA 7.12.2 layouts, wave32) ----
// A: 16x32 (MxK), row-major source [.., lda]. lane 0-15: M=lane, K={0..7,16..23};
// lane 16-31: M=lane-16, K={8..15,24..31}. Pairs packed per VGPR.
__device__ __forceinline__ v16bf load_a_frag(const __bf16* A, int lda, int m0, int k0, int lane){
  int m     = m0 + (lane & 15);
  int khalf = (lane >> 4) << 3;              // 0 or 8
  const __bf16* row = A + (long)m * lda + k0;
  v16bf a;
#pragma unroll
  for (int j = 0; j < 16; ++j){
    int k = ((j >> 3) << 4) + khalf + (j & 7);  // V0-3 -> K base 0, V4-7 -> K base 16
    a[j] = row[k];
  }
  return a;
}
// B: 32x16 (KxN) where Bmat[k][n] = WT[n][k] and WT is the [out,in] weight.
// lane 0-15: N=lane, K={0..7,16..23} ; lane 16-31: N=lane-16, K={8..15,24..31}.
// => per-lane two runs of 8 bf16 from WT row n (matches A).
__device__ __forceinline__ v16bf load_b_frag(const __bf16* WT, int ldk, int n0, int k0, int lane){
  int n    = n0 + (lane & 15);
  int koff = (lane >> 4) << 3;               // 0 or 8
  const __bf16* src = WT + (long)n * ldk + k0 + koff;
  v16bf b;
#pragma unroll
  for (int j = 0; j < 16; ++j) b[j] = src[((j >> 3) << 4) + (j & 7)];
  return b;
}

// ---------------- weight preparation ----------------
__global__ void k_pad_wih(const float* __restrict__ w, __bf16* __restrict__ o){ // [4096,260]->[4096,288]
  int i = blockIdx.x*blockDim.x + threadIdx.x;
  if (i >= G4_*DP_) return;
  int n = i / DP_, k = i % DP_;
  o[i] = (__bf16)(k < D_ ? w[(long)n*D_ + k] : 0.0f);
}
__global__ void k_cvt_whh(const float* __restrict__ w, __bf16* __restrict__ o){  // [4096,1024]
  int i = blockIdx.x*blockDim.x + threadIdx.x;
  if (i >= G4_*H_) return;
  o[i] = (__bf16)w[i];
}
__global__ void k_pad_outw(const float* __restrict__ w, __bf16* __restrict__ o){ // [50,1024]->[64,1024]
  int i = blockIdx.x*blockDim.x + threadIdx.x;
  if (i >= 64*H_) return;
  int n = i / H_, k = i % H_;
  o[i] = (__bf16)(n < TAGS_ ? w[(long)n*H_ + k] : 0.0f);
}
__global__ void k_init_state(__bf16* __restrict__ h0, float* __restrict__ c){
  int i = blockIdx.x*blockDim.x + threadIdx.x;
  if (i >= B_*H_) return;
  h0[i] = (__bf16)0.0f;
  c[i]  = 0.0f;
}

// ---------------- embeddings + char CNN -> Cb bf16 [5120,288] ----------------
__global__ void k_embed(const int* __restrict__ sent, const int* __restrict__ chars,
                        const float* __restrict__ wemb, const float* __restrict__ cemb,
                        const float* __restrict__ convw, const float* __restrict__ convb,
                        __bf16* __restrict__ Cb){
  __shared__ float red[8][256];
  int r = blockIdx.x;            // flat row = b*P + p
  int e = threadIdx.x;           // 0..255
  // word embedding -> cols [0,256)
  int widx = sent[r];
  Cb[(long)r*DP_ + e] = (__bf16)wemb[(long)widx*E_ + e];
  // char conv partials: 2 windows x 4 channels
  int c0 = chars[r*LC_+0], c1 = chars[r*LC_+1], c2 = chars[r*LC_+2], c3 = chars[r*LC_+3];
  float ce0 = cemb[(long)c0*E_+e], ce1 = cemb[(long)c1*E_+e];
  float ce2 = cemb[(long)c2*E_+e], ce3 = cemb[(long)c3*E_+e];
#pragma unroll
  for (int ch = 0; ch < CH_; ++ch){
    const float* w = convw + ch*(KW_*E_);
    float w0 = w[0*E_+e], w1 = w[1*E_+e], w2 = w[2*E_+e];
    red[ch*2+0][e] = ce0*w0 + ce1*w1 + ce2*w2;   // window 0 (chars 0..2)
    red[ch*2+1][e] = ce1*w0 + ce2*w1 + ce3*w2;   // window 1 (chars 1..3)
  }
  __syncthreads();
  for (int s = 128; s > 0; s >>= 1){
    if (e < s){
#pragma unroll
      for (int q = 0; q < 8; ++q) red[q][e] += red[q][e+s];
    }
    __syncthreads();
  }
  if (e < CH_){
    float v = fmaxf(red[e*2][0], red[e*2+1][0]) + convb[e];   // bias invariant under max
    Cb[(long)r*DP_ + E_ + e] = (__bf16)v;
  }
  if (e >= CH_ && e < (DP_ - E_))                              // zero pad cols 260..287
    Cb[(long)r*DP_ + E_ + e] = (__bf16)0.0f;
}

// ---------------- input projection: Gx = Cb @ w_ih^T + b_ih  -> [5120,4096] f32 ----------------
__global__ void k_gemm_ih(const __bf16* __restrict__ Cb, const __bf16* __restrict__ Wih,
                          const float* __restrict__ b_ih, float* __restrict__ Gx){
  int lane = threadIdx.x & 31;
  int wave = threadIdx.x >> 5;
  int m0 = blockIdx.x << 4;                         // 320 M tiles
  int n0 = (((blockIdx.y << 2) + wave) << 4);       // 64*4 = 256 N tiles
  v8f acc = {0.f,0.f,0.f,0.f,0.f,0.f,0.f,0.f};
#pragma unroll
  for (int k0 = 0; k0 < DP_; k0 += 32){
    v16bf a = load_a_frag(Cb, DP_, m0, k0, lane);
    v16bf b = load_b_frag(Wih, DP_, n0, k0, lane);
    acc = __builtin_amdgcn_wmma_f32_16x16x32_bf16(false, a, false, b, (short)0, acc, false, false);
  }
  int col = n0 + (lane & 15);
  int rb  = m0 + ((lane >> 4) << 3);
  float bias = b_ih[col];
#pragma unroll
  for (int v = 0; v < 8; ++v)
    Gx[(long)(rb + v)*G4_ + col] = acc[v] + bias;
}

// ---------------- one LSTM step: 4 waves/block = gates i,f,g,o on one 16x16 h-tile ----------------
__global__ void k_lstm_step(int t, const __bf16* __restrict__ h_in, const __bf16* __restrict__ Whh,
                            const float* __restrict__ Gx, const float* __restrict__ b_hh,
                            float* __restrict__ c, __bf16* __restrict__ h_out,
                            __bf16* __restrict__ hs){
  __shared__ float gl[4][16][16];
  int lane = threadIdx.x & 31;
  int gate = threadIdx.x >> 5;                 // 0..3 : i,f,g,o
  int m0  = blockIdx.x << 4;                   // batch tile (B=64 -> 4)
  int n0h = blockIdx.y << 4;                   // hidden tile (H=1024 -> 64)
  int gcol0 = gate*H_ + n0h;
  int col = gcol0 + (lane & 15);
  int rb  = m0 + ((lane >> 4) << 3);
  float bias = b_hh[col];
  v8f acc;
#pragma unroll
  for (int v = 0; v < 8; ++v)
    acc[v] = Gx[(long)(t*B_ + rb + v)*G4_ + col] + bias;   // x-projection (incl b_ih) + b_hh
  for (int k0 = 0; k0 < H_; k0 += 32){
    v16bf a = load_a_frag(h_in, H_, m0, k0, lane);
    v16bf b = load_b_frag(Whh, H_, gcol0, k0, lane);
    acc = __builtin_amdgcn_wmma_f32_16x16x32_bf16(false, a, false, b, (short)0, acc, false, false);
  }
  int rr0 = (lane >> 4) << 3;
  int cc  = lane & 15;
#pragma unroll
  for (int v = 0; v < 8; ++v) gl[gate][rr0 + v][cc] = acc[v];
  __syncthreads();
  for (int idx = threadIdx.x; idx < 256; idx += 128){
    int rr = idx >> 4, c2 = idx & 15;
    float ig = sigf(gl[0][rr][c2]);
    float fg = sigf(gl[1][rr][c2]);
    float gg = tanhf(gl[2][rr][c2]);
    float og = sigf(gl[3][rr][c2]);
    int  m  = m0 + rr;
    int  hc = n0h + c2;
    long ci = (long)m*H_ + hc;
    float cn = fg*c[ci] + ig*gg;
    c[ci] = cn;
    float hn = og * tanhf(cn);
    __bf16 hb = (__bf16)hn;
    h_out[ci] = hb;                                // ping-pong h for next step
    hs[(long)(t*B_ + m)*H_ + hc] = hb;             // history for the head GEMM
  }
}

// ---------------- head: T = hs[5120,1024] @ out_w^T (padded to 64) + out_b ----------------
__global__ void k_head(const __bf16* __restrict__ hs, const __bf16* __restrict__ Wo,
                       const float* __restrict__ out_b, float* __restrict__ T){
  int lane = threadIdx.x & 31;
  int wave = threadIdx.x >> 5;
  int m0 = blockIdx.x << 4;
  int n0 = wave << 4;                               // 4 N tiles cover 64 (>=50) cols
  v8f acc = {0.f,0.f,0.f,0.f,0.f,0.f,0.f,0.f};
  for (int k0 = 0; k0 < H_; k0 += 32){
    v16bf a = load_a_frag(hs, H_, m0, k0, lane);
    v16bf b = load_b_frag(Wo, H_, n0, k0, lane);
    acc = __builtin_amdgcn_wmma_f32_16x16x32_bf16(false, a, false, b, (short)0, acc, false, false);
  }
  int col = n0 + (lane & 15);
  int rb  = m0 + ((lane >> 4) << 3);
  float bias = (col < TAGS_) ? out_b[col] : 0.0f;
#pragma unroll
  for (int v = 0; v < 8; ++v)
    T[(long)(rb + v)*64 + col] = acc[v] + bias;
}

// ---------------- log_softmax over the P axis (per (b, tag)) ----------------
__global__ void k_logsoftmax(const float* __restrict__ T, float* __restrict__ out){
  int b   = blockIdx.x;        // 0..63
  int tag = threadIdx.x;       // 0..63 (only <50 active)
  if (tag >= TAGS_) return;
  float m = -3.0e30f;
  for (int p = 0; p < P_; ++p) m = fmaxf(m, T[(long)(b*P_ + p)*64 + tag]);
  float s = 0.0f;
  for (int p = 0; p < P_; ++p) s += __expf(T[(long)(b*P_ + p)*64 + tag] - m);
  float lse = m + __logf(s);
  for (int p = 0; p < P_; ++p)
    out[(long)(b*P_ + p)*TAGS_ + tag] = T[(long)(b*P_ + p)*64 + tag] - lse;
}

// ---------------- host driver ----------------
extern "C" void kernel_launch(void* const* d_in, const int* in_sizes, int n_in,
                              void* d_out, int out_size, void* d_ws, size_t ws_size,
                              hipStream_t stream){
  (void)in_sizes; (void)n_in; (void)out_size; (void)ws_size;
  const int*   sent  = (const int*)  d_in[0];
  const int*   chars = (const int*)  d_in[1];
  const float* wemb  = (const float*)d_in[2];
  const float* cemb  = (const float*)d_in[3];
  const float* convw = (const float*)d_in[4];
  const float* convb = (const float*)d_in[5];
  const float* w_ih  = (const float*)d_in[6];
  const float* w_hh  = (const float*)d_in[7];
  const float* b_ih  = (const float*)d_in[8];
  const float* b_hh  = (const float*)d_in[9];
  const float* out_w = (const float*)d_in[10];
  const float* out_b = (const float*)d_in[11];
  float* out = (float*)d_out;

  char* ws = (char*)d_ws;
  size_t off = 0;
  auto alloc = [&](size_t bytes)->char* {
    char* p = ws + off; off += (bytes + 255) & ~(size_t)255; return p;
  };
  __bf16* WihB  = (__bf16*)alloc((size_t)G4_*DP_*2);     // 2.36 MB
  __bf16* WhhB  = (__bf16*)alloc((size_t)G4_*H_*2);      // 8.39 MB (L2-resident across 80 steps)
  __bf16* WoB   = (__bf16*)alloc((size_t)64*H_*2);       // 128 KB
  __bf16* Cb    = (__bf16*)alloc((size_t)ROWS_*DP_*2);   // 2.95 MB
  float*  Gx    = (float*) alloc((size_t)ROWS_*G4_*4);   // 83.9 MB
  __bf16* hsB   = (__bf16*)alloc((size_t)ROWS_*H_*2);    // 10.5 MB
  float*  cst   = (float*) alloc((size_t)B_*H_*4);       // 256 KB
  __bf16* hping = (__bf16*)alloc((size_t)B_*H_*2);       // 128 KB
  __bf16* hpong = (__bf16*)alloc((size_t)B_*H_*2);       // 128 KB
  float*  Tt    = (float*) alloc((size_t)ROWS_*64*4);    // 1.31 MB

  k_pad_wih   <<<(G4_*DP_ + 255)/256, 256, 0, stream>>>(w_ih,  WihB);
  k_cvt_whh   <<<(G4_*H_  + 255)/256, 256, 0, stream>>>(w_hh,  WhhB);
  k_pad_outw  <<<(64*H_   + 255)/256, 256, 0, stream>>>(out_w, WoB);
  k_init_state<<<(B_*H_   + 255)/256, 256, 0, stream>>>(hping, cst);

  k_embed<<<ROWS_, 256, 0, stream>>>(sent, chars, wemb, cemb, convw, convb, Cb);

  // one big WMMA GEMM for the whole input projection (flat reshape is a no-op)
  k_gemm_ih<<<dim3(ROWS_/16, G4_/64), 128, 0, stream>>>(Cb, WihB, b_ih, Gx);

  // sequential recurrence, h ping-ponged in bf16
  for (int t = 0; t < P_; ++t){
    const __bf16* hin = (t & 1) ? hpong : hping;
    __bf16*       hout = (t & 1) ? hping : hpong;
    k_lstm_step<<<dim3(B_/16, H_/16), 128, 0, stream>>>(t, hin, WhhB, Gx, b_hh, cst, hout, hsB);
  }

  k_head<<<ROWS_/16, 128, 0, stream>>>(hsB, WoB, out_b, Tt);
  k_logsoftmax<<<B_, 64, 0, stream>>>(Tt, out);
}
